// Switcher_28157805593225
// MI455X (gfx1250) — compile-verified, matrix operand order checked by reading, not run on hardware
//
#include <hip/hip_runtime.h>

// ---------------------------------------------------------------------------
// Problem constants (match reference)
// ---------------------------------------------------------------------------
#define B_  2
#define T_  2048
#define D_  2048
#define H_  16
#define HD_ 128
#define KW_ 4

typedef __attribute__((ext_vector_type(16))) __bf16 v16bf;
typedef __attribute__((ext_vector_type(8)))  float  v8f;

union Frag16 { uint4 q[2]; v16bf v; };

// ---------------------------------------------------------------------------
// gfx1250 async global->LDS copy (ASYNCcnt-tracked, no VGPR round trip).
// Builtin signature (probe-confirmed via diagnostics):
//   void __builtin_amdgcn_global_load_async_to_lds_b128(
//       v4i addrspace(1)* src, v4i addrspace(3)* dst, imm int off, imm int cpol)
// Guarded: falls back to plain per-lane b128 copy if builtin is absent.
// ---------------------------------------------------------------------------
#if defined(__AMDGCN__) && __has_builtin(__builtin_amdgcn_global_load_async_to_lds_b128)
#define HAVE_ASYNC_LDS 1
#else
#define HAVE_ASYNC_LDS 0
#endif

#if HAVE_ASYNC_LDS
typedef int v4i_ __attribute__((vector_size(16)));
typedef v4i_ __attribute__((address_space(1)))* gbl_v4i_ptr;
typedef v4i_ __attribute__((address_space(3)))* lds_v4i_ptr;
#if __has_builtin(__builtin_amdgcn_s_wait_asynccnt)
#define WAIT_ASYNC() __builtin_amdgcn_s_wait_asynccnt(0)
#else
#define WAIT_ASYNC() asm volatile("s_wait_asynccnt 0" ::: "memory")
#endif
#else
#define WAIT_ASYNC() ((void)0)
#endif

__device__ __forceinline__ void cp_b128(void* lds, const void* g) {
#if HAVE_ASYNC_LDS
  __builtin_amdgcn_global_load_async_to_lds_b128(
      (gbl_v4i_ptr)(g), (lds_v4i_ptr)(lds), /*offset=*/0, /*cpol=*/0);
#else
  *reinterpret_cast<uint4*>(lds) = *reinterpret_cast<const uint4*>(g);
#endif
}

__device__ __forceinline__ v8f zero8() {
  v8f z = {0.f, 0.f, 0.f, 0.f, 0.f, 0.f, 0.f, 0.f};
  return z;
}

__device__ __forceinline__ __bf16 f2bf(float f) {
  unsigned int u = __builtin_bit_cast(unsigned int, f);
  unsigned int r = 0x7FFFu + ((u >> 16) & 1u);
  unsigned short h = (unsigned short)((u + r) >> 16);
  return __builtin_bit_cast(__bf16, h);
}

// Load one 16x32 (MxK) WMMA operand fragment from a row-major LDS tile.
// Per ISA layout: K(lane,j) = (lane/16)*8 + (j/8)*16 + (j%8).
__device__ __forceinline__ v16bf ld_frag(const __bf16* base, int stride, int lane) {
  const int row = lane & 15;
  const int kb  = (lane >> 4) << 3;   // 0 or 8
  const __bf16* p = base + row * stride + kb;
  Frag16 f;
  f.q[0] = *reinterpret_cast<const uint4*>(p);        // K = kb .. kb+7
  f.q[1] = *reinterpret_cast<const uint4*>(p + 16);   // K = kb+16 .. kb+23
  return f.v;
}

__device__ __forceinline__ v8f wmma_bf16(v16bf a, v16bf b, v8f c) {
  return __builtin_amdgcn_wmma_f32_16x16x32_bf16(
      /*neg_a=*/false, a, /*neg_b=*/false, b,
      /*c_mod=*/(short)0, c, /*reuse_a=*/false, /*reuse_b=*/false);
}

// ---------------------------------------------------------------------------
// fp32 -> bf16 conversion (plain)
// ---------------------------------------------------------------------------
__global__ void cvt_f32_bf16(const float* __restrict__ x, __bf16* __restrict__ y,
                             size_t n) {
  size_t i = (size_t)blockIdx.x * blockDim.x + threadIdx.x;
  if (i < n) y[i] = f2bf(x[i]);
}

// ---------------------------------------------------------------------------
// fp32 [R][C] -> bf16 transposed [C][R], LDS-tiled so both sides coalesce.
// grid = (C/32, R/32), block = 256.
// ---------------------------------------------------------------------------
__global__ __launch_bounds__(256) void cvt_transpose_bf16(
    const float* __restrict__ x, __bf16* __restrict__ y, int R, int C)
{
  __shared__ __bf16 tile[32][33];
  const int c0 = blockIdx.x * 32, r0 = blockIdx.y * 32;
  const int lr = threadIdx.x >> 5;        // 0..7
  const int lc = threadIdx.x & 31;        // 0..31
  #pragma unroll
  for (int i = 0; i < 4; ++i)
    tile[lr + i * 8][lc] = f2bf(x[(size_t)(r0 + lr + i * 8) * C + c0 + lc]);
  __syncthreads();
  #pragma unroll
  for (int i = 0; i < 4; ++i)
    y[(size_t)(c0 + lr + i * 8) * R + r0 + lc] = tile[lc][lr + i * 8];
}

// ---------------------------------------------------------------------------
// C[M,N] (f32) = A[M,K] (bf16 row-major) x B[K,N], with B given TRANSPOSED
// as Bt[N][K] (bf16 row-major). 128x64 tile per 256-thread block (8 waves),
// K-chunk 64, double-buffered LDS fed by async global->LDS copies.
// ---------------------------------------------------------------------------
__global__ __launch_bounds__(256) void gemm_bf16_wmma(
    const __bf16* __restrict__ A, const __bf16* __restrict__ Bt,
    float* __restrict__ C, int M, int N, int K)
{
  __shared__ __align__(16) __bf16 As[2][128][72];   // [m][k], k<64 used
  __shared__ __align__(16) __bf16 Bs[2][64][72];    // [n][k], k<64 used
  const int tid = threadIdx.x, lane = tid & 31, wave = tid >> 5;
  const int m0 = blockIdx.x * 128, n0 = blockIdx.y * 64;

  v8f acc[4];
  #pragma unroll
  for (int nt = 0; nt < 4; ++nt) acc[nt] = zero8();

  auto issue = [&](int buf, int k0) {
    // A tile: 128x64 bf16 = 1024 b128 copies, 4 per thread
    #pragma unroll
    for (int i = 0; i < 4; ++i) {
      int idx = tid + i * 256;
      int r = idx >> 3, c = (idx & 7) * 8;
      cp_b128(&As[buf][r][c], A + (size_t)(m0 + r) * K + k0 + c);
    }
    // B tile: 64x64 bf16 = 512 b128 copies, 2 per thread
    #pragma unroll
    for (int i = 0; i < 2; ++i) {
      int idx = tid + i * 256;
      int r = idx >> 3, c = (idx & 7) * 8;
      cp_b128(&Bs[buf][r][c], Bt + (size_t)(n0 + r) * K + k0 + c);
    }
  };

  issue(0, 0);
  int buf = 0;
  for (int k0 = 0; k0 < K; k0 += 64) {
    WAIT_ASYNC();
    __syncthreads();
    if (k0 + 64 < K) issue(buf ^ 1, k0 + 64);   // overlap copy with compute
    #pragma unroll
    for (int kc = 0; kc < 2; ++kc) {
      v16bf a = ld_frag(&As[buf][wave * 16][kc * 32], 72, lane);
      #pragma unroll
      for (int nt = 0; nt < 4; ++nt) {
        v16bf b = ld_frag(&Bs[buf][nt * 16][kc * 32], 72, lane);
        acc[nt] = wmma_bf16(a, b, acc[nt]);
      }
    }
    buf ^= 1;
  }
  // C layout: row = (lane/16)*8 + r, col = lane&15 within each 16x16 tile
  const int rg = (lane >> 4) * 8, cn = lane & 15;
  #pragma unroll
  for (int nt = 0; nt < 4; ++nt)
    #pragma unroll
    for (int r = 0; r < 8; ++r)
      C[(size_t)(m0 + wave * 16 + rg + r) * N + (n0 + nt * 16 + cn)] = acc[nt][r];
}

// ---------------------------------------------------------------------------
// Depthwise causal conv (K=4) + SiLU; writes f32 copy and a bf16 copy that is
// either layout-preserving [B,T,H*HD] or per-head transposed [B,H,HD,T].
// ---------------------------------------------------------------------------
__global__ void conv_silu(const float* __restrict__ x, const float* __restrict__ w,
                          float* __restrict__ y, __bf16* __restrict__ yb,
                          int transposed)
{
  int idx = blockIdx.x * blockDim.x + threadIdx.x;
  if (idx >= B_ * T_ * D_) return;
  int d = idx % D_;
  int t = (idx / D_) % T_;
  int b = idx / (T_ * D_);
  float acc = 0.f;
  #pragma unroll
  for (int i = 0; i < KW_; ++i) {
    int ts = t - (KW_ - 1) + i;
    float xv = (ts >= 0) ? x[idx + (i - (KW_ - 1)) * D_] : 0.f;
    acc += xv * w[d * KW_ + i];
  }
  float s = acc * (1.f / (1.f + __expf(-acc)));   // silu
  y[idx] = s;
  if (transposed) {
    int h = d >> 7, dd = d & 127;
    yb[(((size_t)b * H_ + h) * HD_ + dd) * T_ + t] = f2bf(s);
  } else {
    yb[idx] = f2bf(s);
  }
}

// ---------------------------------------------------------------------------
// beta = sigmoid(hs @ Wb), Wb [D,H]; one block per (b,t) row, 8 lanes/head.
// ---------------------------------------------------------------------------
__global__ __launch_bounds__(128) void beta_proj(
    const float* __restrict__ hs, const float* __restrict__ Wb,
    float* __restrict__ beta)
{
  int row = blockIdx.x;              // 0..B*T-1
  int h    = threadIdx.x >> 3;       // 0..15
  int part = threadIdx.x & 7;
  const float* xr = hs + (size_t)row * D_;
  float acc = 0.f;
  for (int j = part; j < D_; j += 8) acc += xr[j] * Wb[(size_t)j * H_ + h];
  #pragma unroll
  for (int m = 1; m < 8; m <<= 1) acc += __shfl_xor(acc, m, 32);
  if (part == 0) beta[(size_t)row * H_ + h] = 1.f / (1.f + __expf(-acc));
}

// ---------------------------------------------------------------------------
// In-place per-head l2 normalization: x *= rsqrt(sum(x^2)+1e-6) * scale
// grid = B*T*H, block = 128 (=HD)
// ---------------------------------------------------------------------------
__global__ __launch_bounds__(128) void l2norm_heads(float* __restrict__ x, float scale)
{
  size_t base = (size_t)blockIdx.x * HD_;
  float v = x[base + threadIdx.x];
  float ss = v * v;
  #pragma unroll
  for (int m = 1; m < 32; m <<= 1) ss += __shfl_xor(ss, m, 32);
  __shared__ float red[4];
  int wave = threadIdx.x >> 5, lane = threadIdx.x & 31;
  if (lane == 0) red[wave] = ss;
  __syncthreads();
  float tot = red[0] + red[1] + red[2] + red[3];
  x[base + threadIdx.x] = v * rsqrtf(tot + 1e-6f) * scale;
}

// ---------------------------------------------------------------------------
// DeltaNet recurrence. One block per (b,h); thread tid owns state column tid:
// S[k] (128 fp32) lives in registers (fully unrolled). Sequential in T.
// ---------------------------------------------------------------------------
__global__ __launch_bounds__(128) void delta_rule(
    const float* __restrict__ qn, const float* __restrict__ kn,
    const float* __restrict__ v,  const float* __restrict__ beta,
    float* __restrict__ olin)
{
  __shared__ float ks[HD_], qs[HD_], vs[HD_];
  const int tid = threadIdx.x;
  const int bh = blockIdx.x, b = bh / H_, h = bh % H_;
  float S[HD_];
  #pragma unroll
  for (int k = 0; k < HD_; ++k) S[k] = 0.f;

  for (int t = 0; t < T_; ++t) {
    size_t off = (((size_t)b * T_ + t) * H_ + h) * HD_ + tid;
    ks[tid] = kn[off]; qs[tid] = qn[off]; vs[tid] = v[off];
    __syncthreads();
    float bta = beta[((size_t)b * T_ + t) * H_ + h];
    float kv = 0.f;
    #pragma unroll
    for (int k = 0; k < HD_; ++k) kv += ks[k] * S[k];
    float vn = bta * (vs[tid] - kv);
    float o = 0.f;
    #pragma unroll
    for (int k = 0; k < HD_; ++k) { S[k] += ks[k] * vn; o += qs[k] * S[k]; }
    olin[off] = o;
    __syncthreads();
  }
}

// ---------------------------------------------------------------------------
// Causal flash attention (bf16 WMMA, fp32 accumulate, online softmax).
// grid = (T/64, B*H), block = 128 (4 waves, 16 query rows per wave).
// Q,K in [B,T,H,HD]; V pre-transposed to [B,H,HD,T] so all tiles are plain
// row-major async b128 copies (no in-kernel transpose scatter).
// ---------------------------------------------------------------------------
__global__ __launch_bounds__(128) void flash_attn(
    const __bf16* __restrict__ Qb, const __bf16* __restrict__ Kb,
    const __bf16* __restrict__ Vt_g, float* __restrict__ Oa)
{
  __shared__ __align__(16) __bf16 Qs[64][136];  // padded, conflict-free frags
  __shared__ __align__(16) __bf16 Ks[32][136];
  __shared__ __align__(16) __bf16 Vt[128][48];  // [hd][key]
  __shared__ __align__(16) __bf16 Ps[64][48];   // [q-row][key]
  const int tid = threadIdx.x, lane = tid & 31, wave = tid >> 5;
  const int bh = blockIdx.y, b = bh / H_, h = bh % H_;
  const int q0 = blockIdx.x * 64;
  const float sc = 0.08838834764831845f;        // HD^-0.5

  // Q tile: 64x128 bf16 = 1024 b128 copies, 8 per thread
  #pragma unroll
  for (int i = 0; i < 8; ++i) {
    int idx = tid + i * 128;
    int r = idx >> 4, c = (idx & 15) * 8;
    cp_b128(&Qs[r][c],
            Qb + (((size_t)b * T_ + (q0 + r)) * H_ + h) * HD_ + c);
  }

  float mprev[8], lsum[8];
  v8f o[8];
  #pragma unroll
  for (int r = 0; r < 8; ++r) { mprev[r] = -__builtin_inff(); lsum[r] = 0.f; }
  #pragma unroll
  for (int nt = 0; nt < 8; ++nt) o[nt] = zero8();

  const int kblocks = (q0 + 64) / 32;           // causal: keys 0 .. q0+63
  for (int kb = 0; kb < kblocks; ++kb) {
    __syncthreads();                            // prior reads of Ks/Vt done
    // K tile 32x128: 4 copies/thread
    #pragma unroll
    for (int i = 0; i < 4; ++i) {
      int idx = tid + i * 128;
      int r = idx >> 4, c = (idx & 15) * 8;
      cp_b128(&Ks[r][c],
              Kb + (((size_t)b * T_ + (kb * 32 + r)) * H_ + h) * HD_ + c);
    }
    // V tile (already transposed in global): 128x32: 4 copies/thread
    #pragma unroll
    for (int i = 0; i < 4; ++i) {
      int idx = tid + i * 128;
      int r = idx >> 2, c = (idx & 3) * 8;
      cp_b128(&Vt[r][c],
              Vt_g + (((size_t)b * H_ + h) * HD_ + r) * T_ + kb * 32 + c);
    }
    WAIT_ASYNC();
    __syncthreads();

    // Scores: [16 q-rows x 32 keys] per wave over HD=128 in 4 K-chunks
    v8f s0 = zero8(), s1 = zero8();
    #pragma unroll
    for (int kc = 0; kc < 4; ++kc) {
      v16bf aq = ld_frag(&Qs[wave * 16][kc * 32], 136, lane);
      v16bf b0 = ld_frag(&Ks[0][kc * 32],  136, lane);
      v16bf b1 = ld_frag(&Ks[16][kc * 32], 136, lane);
      s0 = wmma_bf16(aq, b0, s0);
      s1 = wmma_bf16(aq, b1, s1);
    }

    // Online softmax (per-lane: 8 rows; 16-lane xor reductions stay in-row)
    const int colg = kb * 32 + (lane & 15);
    const int rowb = q0 + wave * 16 + (lane >> 4) * 8;
    #pragma unroll
    for (int r = 0; r < 8; ++r) {
      int row = rowb + r;
      float v0 = s0[r] * sc; if (colg > row)      v0 = -__builtin_inff();
      float v1 = s1[r] * sc; if (colg + 16 > row) v1 = -__builtin_inff();
      float mx = fmaxf(v0, v1);
      for (int m = 1; m < 16; m <<= 1) mx = fmaxf(mx, __shfl_xor(mx, m, 32));
      float mnew = fmaxf(mprev[r], mx);
      float p0 = __expf(v0 - mnew);
      float p1 = __expf(v1 - mnew);
      float rs = p0 + p1;
      for (int m = 1; m < 16; m <<= 1) rs += __shfl_xor(rs, m, 32);
      float rescale = __expf(mprev[r] - mnew);
      lsum[r] = lsum[r] * rescale + rs;
      mprev[r] = mnew;
      #pragma unroll
      for (int nt = 0; nt < 8; ++nt) o[nt][r] *= rescale;
      int pr = wave * 16 + (lane >> 4) * 8 + r;
      Ps[pr][lane & 15]        = f2bf(p0);
      Ps[pr][16 + (lane & 15)] = f2bf(p1);
    }
    __syncthreads();

    // O += P[16x32] @ V[32x128]  (8 WMMA per wave)
    v16bf ap = ld_frag(&Ps[wave * 16][0], 48, lane);
    #pragma unroll
    for (int nt = 0; nt < 8; ++nt) {
      v16bf bv = ld_frag(&Vt[nt * 16][0], 48, lane);
      o[nt] = wmma_bf16(ap, bv, o[nt]);
    }
  }

  // Epilogue: o / l, fp32 out [B,T,H,HD]
  #pragma unroll
  for (int nt = 0; nt < 8; ++nt)
    #pragma unroll
    for (int r = 0; r < 8; ++r) {
      int t = q0 + wave * 16 + (lane >> 4) * 8 + r;
      int d = nt * 16 + (lane & 15);
      Oa[(((size_t)b * T_ + t) * H_ + h) * HD_ + d] = o[nt][r] / lsum[r];
    }
}

// ---------------------------------------------------------------------------
// rmsnorm both branches, mix by s_mask, gate by sigmoid(gproj);
// write bf16 mixed activations + copy s_mask to output tail.
// grid = B*T*H, block = 128
// ---------------------------------------------------------------------------
__global__ __launch_bounds__(128) void mix_norm_gate(
    const float* __restrict__ olin, const float* __restrict__ oattn,
    const float* __restrict__ gproj, const float* __restrict__ onw,
    const float* __restrict__ smask, __bf16* __restrict__ mixed,
    float* __restrict__ outTail)
{
  int bh = blockIdx.x;
  size_t base = (size_t)bh * HD_;
  int tid = threadIdx.x;
  float la = oattn[base + tid], ll = olin[base + tid];
  float sa = la * la, sl = ll * ll;
  #pragma unroll
  for (int m = 1; m < 32; m <<= 1) {
    sa += __shfl_xor(sa, m, 32);
    sl += __shfl_xor(sl, m, 32);
  }
  __shared__ float ra[4], rl[4];
  int wave = tid >> 5, lane = tid & 31;
  if (lane == 0) { ra[wave] = sa; rl[wave] = sl; }
  __syncthreads();
  float ta = ra[0] + ra[1] + ra[2] + ra[3];
  float tl = rl[0] + rl[1] + rl[2] + rl[3];
  float w  = onw[tid];
  float na = la * rsqrtf(ta * (1.f / HD_) + 1e-5f) * w;
  float nl = ll * rsqrtf(tl * (1.f / HD_) + 1e-5f) * w;
  float s  = smask[bh];
  float ov = (1.f - s) * nl + s * na;
  float gp = gproj[base + tid];
  float g  = 1.f / (1.f + __expf(-gp));
  mixed[base + tid] = f2bf(ov * g);
  if (tid == 0) outTail[bh] = s;
}

// ---------------------------------------------------------------------------
// Host launch
// ---------------------------------------------------------------------------
extern "C" void kernel_launch(void* const* d_in, const int* in_sizes, int n_in,
                              void* d_out, int out_size, void* d_ws, size_t ws_size,
                              hipStream_t stream) {
  (void)in_sizes; (void)n_in; (void)out_size; (void)ws_size;
  const float* hs  = (const float*)d_in[0];
  const float* Wq  = (const float*)d_in[1];
  const float* Wk  = (const float*)d_in[2];
  const float* Wv  = (const float*)d_in[3];
  const float* Wo  = (const float*)d_in[4];
  const float* Wg  = (const float*)d_in[5];
  const float* Wb  = (const float*)d_in[6];
  const float* cq  = (const float*)d_in[7];
  const float* ck  = (const float*)d_in[8];
  const float* cv  = (const float*)d_in[9];
  const float* onw = (const float*)d_in[10];
  const float* sm  = (const float*)d_in[11];

  const size_t BTD = (size_t)B_ * T_ * D_;   // 8,388,608
  const size_t DD  = (size_t)D_ * D_;
  const size_t BTH = (size_t)B_ * T_ * H_;

  char* ws = (char*)d_ws;
  size_t off = 0;
  auto take = [&](size_t bytes) -> char* {
    char* p = ws + off;
    off += (bytes + 255) & ~(size_t)255;
    return p;
  };
  __bf16* hsb   = (__bf16*)take(BTD * 2);
  __bf16* Wqt   = (__bf16*)take(DD * 2);   // transposed bf16 weights [N][K]
  __bf16* Wkt   = (__bf16*)take(DD * 2);
  __bf16* Wvt   = (__bf16*)take(DD * 2);
  __bf16* Wgt   = (__bf16*)take(DD * 2);
  __bf16* Wot   = (__bf16*)take(DD * 2);
  float*  proj  = (float*)take(BTD * 4);   // reused for q,k,v projections
  float*  gproj = (float*)take(BTD * 4);
  float*  qc    = (float*)take(BTD * 4);   // conv outputs (l2-normed in place)
  float*  kc    = (float*)take(BTD * 4);
  float*  vc    = (float*)take(BTD * 4);
  __bf16* qcb   = (__bf16*)take(BTD * 2);  // bf16 conv outputs for attention
  __bf16* kcb   = (__bf16*)take(BTD * 2);
  __bf16* vcbT  = (__bf16*)take(BTD * 2);  // V in [B,H,HD,T]
  float*  beta  = (float*)take(BTH * 4);
  float*  olin  = (float*)take(BTD * 4);
  float*  oattn = (float*)take(BTD * 4);
  __bf16* mixed = (__bf16*)take(BTD * 2);

  float* out  = (float*)d_out;
  float* tail = out + BTD;

  // 1) conversions: hs plain; weights bf16 + transposed (WMMA B-operand form)
  cvt_f32_bf16<<<(unsigned)((BTD + 255) / 256), 256, 0, stream>>>(hs, hsb, BTD);
  const dim3 tgrid(D_ / 32, D_ / 32);
  cvt_transpose_bf16<<<tgrid, 256, 0, stream>>>(Wq, Wqt, D_, D_);
  cvt_transpose_bf16<<<tgrid, 256, 0, stream>>>(Wk, Wkt, D_, D_);
  cvt_transpose_bf16<<<tgrid, 256, 0, stream>>>(Wv, Wvt, D_, D_);
  cvt_transpose_bf16<<<tgrid, 256, 0, stream>>>(Wg, Wgt, D_, D_);
  cvt_transpose_bf16<<<tgrid, 256, 0, stream>>>(Wo, Wot, D_, D_);

  const dim3 ggrid((B_ * T_) / 128, D_ / 64);  // (32, 32)
  const unsigned cgrid = (unsigned)((BTD + 255) / 256);

  // 2) projections + conv+silu (projection buffer reused)
  gemm_bf16_wmma<<<ggrid, 256, 0, stream>>>(hsb, Wqt, proj, B_ * T_, D_, D_);
  conv_silu<<<cgrid, 256, 0, stream>>>(proj, cq, qc, qcb, 0);
  gemm_bf16_wmma<<<ggrid, 256, 0, stream>>>(hsb, Wkt, proj, B_ * T_, D_, D_);
  conv_silu<<<cgrid, 256, 0, stream>>>(proj, ck, kc, kcb, 0);
  gemm_bf16_wmma<<<ggrid, 256, 0, stream>>>(hsb, Wvt, proj, B_ * T_, D_, D_);
  conv_silu<<<cgrid, 256, 0, stream>>>(proj, cv, vc, vcbT, 1);
  gemm_bf16_wmma<<<ggrid, 256, 0, stream>>>(hsb, Wgt, gproj, B_ * T_, D_, D_);

  // 3) beta
  beta_proj<<<B_ * T_, 128, 0, stream>>>(hs, Wb, beta);

  // 4) attention branch (raw conv q/k/v, bf16) — before in-place l2norm
  flash_attn<<<dim3(T_ / 64, B_ * H_), 128, 0, stream>>>(qcb, kcb, vcbT, oattn);

  // 5) delta-rule branch: l2norm q (with scale) and k, then recurrence
  l2norm_heads<<<(unsigned)BTH, 128, 0, stream>>>(qc, 0.08838834764831845f);
  l2norm_heads<<<(unsigned)BTH, 128, 0, stream>>>(kc, 1.0f);
  delta_rule<<<B_ * H_, 128, 0, stream>>>(qc, kc, vc, beta, olin);

  // 6) rmsnorm + switch-mix + gate (+ s_mask passthrough to output tail)
  mix_norm_gate<<<(unsigned)BTH, 128, 0, stream>>>(olin, oattn, gproj, onw, sm,
                                                   mixed, tail);

  // 7) output projection straight into d_out
  gemm_bf16_wmma<<<ggrid, 256, 0, stream>>>(mixed, Wot, out, B_ * T_, D_, D_);
}
